// SparkNetAlpha_76922864272044
// MI455X (gfx1250) — matrix operand order repaired and checked by reading, not compiled, not run on hardware
//
#include <hip/hip_runtime.h>
#include <stdint.h>

#define N 8192
#define NSPARK 64
#define ROWS_PER_WG 128        // 8 waves * 16 rows
#define KC 32                  // K tile per LDS stage (1 WMMA step)
#define LDS_STRIDE 36          // KC + 4 pad floats: 144B rows, conflict-free, 16B aligned
#define TILE_FLOATS (ROWS_PER_WG * LDS_STRIDE)
#define KSPLIT 4
#define KRANGE (N / KSPLIT)    // 2048 columns per blockIdx.y
#define NITER (KRANGE / KC)    // 64 double-buffered stages

#define STATE_DECAY 0.95f
#define MEM_DECAY 0.92f
#define NOISE_STD 0.05f
#define LR_EDGE 0.05f
#define GDECAY 0.999f          // 1 - LR_GLOBAL_DECAY
#define INV_GDECAY (1.0f / 0.999f)
#define MEM_BIAS 0.8f
#define INV_TEMP (1.0f / 0.3f)
#define SAT 0.99f
#define EXPLORE 0.05f
#define EN_DECAY 0.98f
#define EN_MIN 0.05f
#define MEM_DEPOSIT 0.15f
#define FORCE_STEPS 5

typedef __attribute__((ext_vector_type(16))) _Float16 v16h;
typedef __attribute__((ext_vector_type(8)))  float    v8f;

// ---- CDNA5 data-mover feature detection (compile-safe fallbacks) ----
#if __has_builtin(__builtin_amdgcn_tensor_load_to_lds) && __has_builtin(__builtin_amdgcn_s_wait_tensorcnt)
#define HAVE_TDM 1
#endif
#if !defined(HAVE_TDM) && __has_builtin(__builtin_amdgcn_global_load_async_to_lds_b128) && __has_builtin(__builtin_amdgcn_s_wait_asynccnt)
#define HAVE_ASYNC 1
#endif

#if defined(HAVE_TDM)
typedef unsigned int tdm_v4u __attribute__((ext_vector_type(4)));
typedef int          tdm_v8i __attribute__((ext_vector_type(8)));
typedef int          tdm_v4i __attribute__((ext_vector_type(4)));
typedef __attribute__((address_space(3))) const char lds_char_t;
#define STAGE_WAIT_ONE()  __builtin_amdgcn_s_wait_tensorcnt(1)
#define STAGE_WAIT_ZERO() __builtin_amdgcn_s_wait_tensorcnt(0)
#elif defined(HAVE_ASYNC)
typedef __attribute__((address_space(1))) const void as1c_void;
typedef __attribute__((address_space(3))) void       as3_void;
#define STAGE_WAIT_ONE()  __builtin_amdgcn_s_wait_asynccnt(4)   // 4 b128 copies per tile per lane
#define STAGE_WAIT_ZERO() __builtin_amdgcn_s_wait_asynccnt(0)
#else
#define STAGE_WAIT_ONE()  ((void)0)
#define STAGE_WAIT_ZERO() ((void)0)
#endif

// ---- deterministic counter-hash RNG helpers ----
__device__ __forceinline__ uint32_t h32(uint32_t x) {
  x ^= x >> 16; x *= 0x7feb352du;
  x ^= x >> 15; x *= 0x846ca68bu;
  x ^= x >> 16; return x;
}
__device__ __forceinline__ float u01(uint32_t h) {
  return ((h >> 8) + 0.5f) * (1.0f / 16777216.0f);   // (0,1)
}
__device__ __forceinline__ float gumbel(uint32_t h) {
  float u = u01(h);
  return -__logf(-__logf(u));
}

#if defined(HAVE_TDM)
__device__ __forceinline__ unsigned lds_addr_of(const void* p) {
  return (unsigned)(unsigned long long)(lds_char_t*)p;
}
#endif

// Issue copy of one ROWS_PER_WG x KC fp32 tile (global row stride N) into padded LDS buffer.
__device__ __forceinline__ void stage_issue(const float* __restrict__ Win, float* lbuf,
                                            int rowBase, int col0, int tid, int wave_u) {
#if defined(HAVE_TDM)
  // One TDM descriptor per wave: 16 rows x 32 cols, engine inserts the 4-DW LDS row pad.
  unsigned long long ga = (unsigned long long)(const void*)
      (Win + (size_t)(rowBase + wave_u * 16) * N + col0);
  unsigned la = lds_addr_of(lbuf) + (unsigned)(wave_u * 16 * LDS_STRIDE * 4);
  tdm_v4u g0;
  g0[0] = 1u;                                                  // count=1 (user descriptor)
  g0[1] = la;                                                  // lds_addr (bytes)
  g0[2] = (unsigned)ga;                                        // global_addr[31:0]
  g0[3] = (unsigned)((ga >> 32) & 0x01FFFFFFu) | (2u << 30);   // global_addr[56:32] | type=2
  tdm_v8i g1;
  g1[0] = (int)((2u << 16)      // data_size = 4B
              | (1u << 20)      // pad_enable
              | (4u << 22)      // pad_interval: every 32 DWORDs
              | (3u << 25));    // pad_amount: 4 DWORDs
  g1[1] = (int)(((unsigned)N & 0xFFFFu) << 16);                              // tensor_dim0 lo16
  g1[2] = (int)((((unsigned)N >> 16) & 0xFFFFu) | (((unsigned)N & 0xFFFFu) << 16)); // dim0 hi | dim1 lo
  g1[3] = (int)((((unsigned)N >> 16) & 0xFFFFu) | ((unsigned)KC << 16));     // dim1 hi | tile_dim0
  g1[4] = 16;                                                  // tile_dim1=16 rows, tile_dim2=0
  g1[5] = N;                                                   // tensor_dim0_stride[31:0]
  g1[6] = 0;
  g1[7] = 0;
  tdm_v4i gz4; gz4[0] = 0; gz4[1] = 0; gz4[2] = 0; gz4[3] = 0; // 2D: groups 2/3 unused
  tdm_v8i gz8; gz8[0] = 0; gz8[1] = 0; gz8[2] = 0; gz8[3] = 0;
               gz8[4] = 0; gz8[5] = 0; gz8[6] = 0; gz8[7] = 0;
  // clang-23 / therock 6-arg form: (g0, g1, g2, g3, <int32x8>, cpol)
  __builtin_amdgcn_tensor_load_to_lds(g0, g1, gz4, gz4, gz8, 0);
#elif defined(HAVE_ASYNC)
  (void)wave_u;
#pragma unroll
  for (int p = 0; p < 4; ++p) {                                // 1024 float4s / 256 threads
    int t4 = tid + p * 256;
    int r  = t4 >> 3;                                          // 8 float4 per row (KC=32)
    int c4 = (t4 & 7) << 2;
    const float* gp = Win + (size_t)(rowBase + r) * N + col0 + c4;
    float* lp = lbuf + r * LDS_STRIDE + c4;
    __builtin_amdgcn_global_load_async_to_lds_b128((as1c_void*)gp, (as3_void*)lp, 0, 0);
  }
#else
  (void)wave_u;
  float4 v[4];
#pragma unroll
  for (int p = 0; p < 4; ++p) {                                // issue all loads first (MLP)
    int t4 = tid + p * 256;
    v[p] = *(const float4*)(Win + (size_t)(rowBase + (t4 >> 3)) * N + col0 + ((t4 & 7) << 2));
  }
#pragma unroll
  for (int p = 0; p < 4; ++p) {
    int t4 = tid + p * 256;
    *(float4*)(lbuf + (t4 >> 3) * LDS_STRIDE + ((t4 & 7) << 2)) = v[p];
  }
#endif
}

// ---- K0: zero y partials, build f16 copy of decayed state ----
__global__ void k_init(const float* __restrict__ s_in,
                       _Float16* __restrict__ xh, float* __restrict__ y) {
  int i = blockIdx.x * 256 + threadIdx.x;
  if (i < N) {
    xh[i] = (_Float16)(s_in[i] * STATE_DECAY);
#pragma unroll
    for (int j = 0; j < KSPLIT; ++j) y[i + j * N] = 0.f;
  }
}

// ---- K1: double-buffered DMA pipeline: y_partial = W @ x (WMMA) + Wout = clip(W*0.999) ----
__global__ void __launch_bounds__(256)
k_gemv_decay(const float* __restrict__ Win, float* __restrict__ Wout,
             const _Float16* __restrict__ xh, float* __restrict__ y) {
  __shared__ float    tile[2 * TILE_FLOATS];   // double-buffered padded W tiles (36 KB)
  __shared__ _Float16 bslice[KRANGE];          // column-vector slice in f16 (4 KB)

  const int tid  = threadIdx.x;
  const int lane = tid & 31;
  const int wave = tid >> 5;                                     // 0..7
  const int wave_u = __builtin_amdgcn_readfirstlane(tid) >> 5;   // provably uniform copy
  const int rowBase = blockIdx.x * ROWS_PER_WG;
  const int colBase = blockIdx.y * KRANGE;

  const int m      = lane & 15;                 // A-matrix row within wave tile
  const int hiHalf = lane >> 4;                 // 0: lanes 0-15, 1: lanes 16-31
  const int kbA    = hiHalf ? 8 : 0;            // 16-bit A layout K base
  const int kbB    = hiHalf ? 8 : 0;            // 16-bit B layout K base
  const int waveRow = wave * 16 + m;

  // stage B slice to LDS once (removes vmem wait from the WMMA critical path)
  for (int i = tid; i < KRANGE; i += 256) bslice[i] = xh[colBase + i];

  v8f acc = {};

  // prologue: start DMA of tile 0
  stage_issue(Win, tile, rowBase, colBase, tid, wave_u);

  for (int it = 0; it < NITER; ++it) {
    const int kk = it * KC;
    float* cur = tile + (it & 1) * TILE_FLOATS;
    if (it + 1 < NITER) {   // uniform branch: EXEC stays all-ones for WMMA below
      stage_issue(Win, tile + ((it + 1) & 1) * TILE_FLOATS,
                  rowBase, colBase + kk + KC, tid, wave_u);
      STAGE_WAIT_ONE();     // only tile `it` must be complete; next DMA runs under compute
    } else {
      STAGE_WAIT_ZERO();
    }
    __syncthreads();

    // decayed/clamped write-back straight from LDS (HBM: exactly 1 read + 1 write of W)
#pragma unroll
    for (int p = 0; p < 4; ++p) {
      int t4 = tid + p * 256;
      int r  = t4 >> 3;
      int c4 = (t4 & 7) << 2;
      const float4 w4 = *(const float4*)(cur + r * LDS_STRIDE + c4);
      float4 d;
      d.x = fminf(fmaxf(w4.x * GDECAY, -1.f), 1.f);
      d.y = fminf(fmaxf(w4.y * GDECAY, -1.f), 1.f);
      d.z = fminf(fmaxf(w4.z * GDECAY, -1.f), 1.f);
      d.w = fminf(fmaxf(w4.w * GDECAY, -1.f), 1.f);
      *(float4*)(Wout + (size_t)(rowBase + r) * N + colBase + kk + c4) = d;
    }

    // one 16x16x32 f16 WMMA per wave per tile, fp32 accumulation
    {
      v16h a, b;
      const float* arow = cur + waveRow * LDS_STRIDE + kbA;
#pragma unroll
      for (int i = 0; i < 8; ++i) {
        a[i]     = (_Float16)arow[i];            // K = kbA + 0..7
        a[i + 8] = (_Float16)arow[i + 16];       // K = kbA + 16..23
      }
      const _Float16* bp = bslice + kk + kbB;
#pragma unroll
      for (int i = 0; i < 16; ++i) b[i] = bp[i < 8 ? i : i + 8]; // B column-replicated state vector
      acc = __builtin_amdgcn_wmma_f32_16x16x32_f16(
          false, a, false, b, (short)0, acc, false, false);
    }
    __syncthreads();   // protect both LDS buffers before next DMA overwrites
  }

  // D layout: lane 0 holds column 0 of rows M=0..7; lane 16 of rows M=8..15.
  if ((lane & 15) == 0) {
    int r0 = rowBase + wave * 16 + (hiHalf ? 8 : 0);
    float* yp = y + (size_t)blockIdx.y * N;      // disjoint partial slice: deterministic
#pragma unroll
    for (int i = 0; i < 8; ++i) yp[r0 + i] = acc[i];
  }
}

// ---- K2: s_out = sigmoid(sum(y partials) + noise), M_out = M*0.92 ----
__global__ void k_activate(const float* __restrict__ y, const float* __restrict__ M_in,
                           float* __restrict__ s_out, float* __restrict__ M_out) {
  int i = blockIdx.x * 256 + threadIdx.x;
  if (i >= N) return;
  float v = 0.f;
#pragma unroll
  for (int j = 0; j < KSPLIT; ++j) v += y[i + j * N];
  float u1 = u01(h32(0x9E3779B9u ^ (uint32_t)i));
  float u2 = u01(h32(0x85EBCA6Bu + (uint32_t)i * 0x27D4EB2Fu));
  float g  = __builtin_sqrtf(-2.0f * __logf(u1)) * __cosf(6.2831853f * u2);
  v += NOISE_STD * g;
  s_out[i] = 1.0f / (1.0f + __expf(-v));
  M_out[i] = M_in[i] * MEM_DECAY;
}

// ---- K3: sequential 64-step spark scan (single workgroup) ----
__global__ void __launch_bounds__(256)
k_scan(float* __restrict__ W, float* __restrict__ s, float* __restrict__ M,
       const int* __restrict__ pos_in, const float* __restrict__ en_in,
       const int* __restrict__ age_in,
       int* __restrict__ pos_out, float* __restrict__ en_out, int* __restrict__ age_out) {
  __shared__ float rv[256];  __shared__ int ri[256];
  __shared__ float rv2[256]; __shared__ int ri2[256];
  const int tid = threadIdx.x;

  // force young sparks to full activation (sigmoid < 1, so set == max; dups idempotent)
  if (tid < NSPARK) {
    if (age_in[tid] < FORCE_STEPS) s[pos_in[tid]] = 1.0f;
  }
  __threadfence_block();
  __syncthreads();

  for (int step = 0; step < NSPARK; ++step) {
    const int   prev   = pos_in[step];
    float       energy = en_in[step];
    const int   age    = age_in[step];
    const float s_prev = s[prev];

    // Gumbel-max categorical over transition logits, and over memory dist (respawn)
    float best = -1e30f;  int bi = 0;
    float best2 = -1e30f; int bi2 = 0;
    const uint32_t salt = 0xC0FFEEu + (uint32_t)step * 0x01000193u;
    for (int i = tid; i < N; i += 256) {
      float w    = W[(size_t)prev * N + i] * INV_GDECAY;  // undo fused pre-decay
      float base = (w > 0.f ? w : 0.f) + 1e-6f;
      float m    = M[i];
      float logit = base * INV_TEMP + MEM_BIAS * m;
      if (s[i] >= SAT) logit = -1e9f;
      float v = logit + gumbel(h32(salt ^ (uint32_t)i));
      if (v > best) { best = v; bi = i; }
      float v2 = 2.0f * m + gumbel(h32((salt * 0x9E3779B9u) ^ (uint32_t)(i + N)));
      if (v2 > best2) { best2 = v2; bi2 = i; }
    }
    rv[tid] = best;  ri[tid] = bi;
    rv2[tid] = best2; ri2[tid] = bi2;
    __syncthreads();
    for (int off = 128; off > 0; off >>= 1) {
      if (tid < off) {
        if (rv[tid + off] > rv[tid])   { rv[tid] = rv[tid + off];   ri[tid] = ri[tid + off]; }
        if (rv2[tid + off] > rv2[tid]) { rv2[tid] = rv2[tid + off]; ri2[tid] = ri2[tid + off]; }
      }
      __syncthreads();
    }

    if (tid == 0) {
      const int samp = ri[0];
      const int memp = ri2[0];
      uint32_t he = h32(0xA5A5F00Du + (uint32_t)step * 0x9E3779B9u);
      bool explore = u01(he) < EXPLORE;
      int  randp   = (int)(h32(he ^ 0x7F4A7C15u) & (N - 1));
      int  nxt     = explore ? randp : samp;

      // Hebbian update on pre-decay value, re-apply decay+clip (stored form is decayed)
      float w_old = W[(size_t)nxt * N + prev] * INV_GDECAY;
      float w_new = w_old * (1.0f - LR_EDGE) + s_prev * LR_EDGE;
      w_new = fminf(fmaxf(w_new * GDECAY, -1.f), 1.f);
      W[(size_t)nxt * N + prev] = w_new;

      M[nxt] += MEM_DEPOSIT;
      energy *= EN_DECAY;
      s[nxt] = energy;

      // respawn affects outputs only (reference semantics); M.sum() > 0 always holds
      bool dead = energy < EN_MIN;
      pos_out[step] = dead ? memp : nxt;
      en_out[step]  = dead ? 1.0f : energy;
      age_out[step] = dead ? 0 : (age + 1);
    }
    __threadfence_block();
    __syncthreads();
  }
}

extern "C" void kernel_launch(void* const* d_in, const int* in_sizes, int n_in,
                              void* d_out, int out_size, void* d_ws, size_t ws_size,
                              hipStream_t stream) {
  (void)in_sizes; (void)n_in; (void)out_size; (void)ws_size;
  const float* Win    = (const float*)d_in[0];
  const float* s_in   = (const float*)d_in[1];
  const float* M_in   = (const float*)d_in[2];
  const int*   pos_in = (const int*)d_in[3];
  const float* en_in  = (const float*)d_in[4];
  const int*   age_in = (const int*)d_in[5];

  float* Wout   = (float*)d_out;
  float* s_out  = Wout + (size_t)N * N;
  float* M_out  = s_out + N;
  int*   pos_o  = (int*)(M_out + N);
  float* en_o   = (float*)(pos_o + NSPARK);
  int*   age_o  = (int*)(en_o + NSPARK);

  float*    y  = (float*)d_ws;                    // KSPLIT*N fp32 partials (no atomics)
  _Float16* xh = (_Float16*)(y + KSPLIT * N);     // N f16 copy of decayed state

  k_init<<<N / 256, 256, 0, stream>>>(s_in, xh, y);

  dim3 g1(N / ROWS_PER_WG, KSPLIT);               // 64 x 4 workgroups, 8 waves each
  k_gemv_decay<<<g1, 256, 0, stream>>>(Win, Wout, xh, y);

  k_activate<<<N / 256, 256, 0, stream>>>(y, M_in, s_out, M_out);

  k_scan<<<1, 256, 0, stream>>>(Wout, s_out, M_out, pos_in, en_in, age_in,
                                pos_o, en_o, age_o);
}
